// Attention_16217796510290
// MI455X (gfx1250) — matrix operand order repaired and checked by reading, not compiled, not run on hardware
//
#include <hip/hip_runtime.h>
#include <math.h>
#include <stdint.h>

typedef __bf16 bf16;
typedef __attribute__((ext_vector_type(16))) __bf16 v16bf;
typedef __attribute__((ext_vector_type(8)))  float  v8f;

// ---------------------------------------------------------------------------
// WMMA helpers (CDNA5 wave32, D = A(16x32) * B(32x16) + C, bf16 in / f32 acc)
// ---------------------------------------------------------------------------
__device__ __forceinline__ v8f wmma_bf16(v16bf a, v16bf b, v8f c) {
  return __builtin_amdgcn_wmma_f32_16x16x32_bf16(
      /*neg_a=*/false, a, /*neg_b=*/false, b,
      /*c_mod=*/(short)0, c, /*reuse_a=*/false, /*reuse_b=*/false);
}

// A-matrix 16x32 fragment: lane L holds row m=L&15; element i -> K index
// k0 + (L>>4)*16 + i
__device__ __forceinline__ v16bf load_a_frag(const bf16* s, int row0, int k0, int ldk) {
  int lane = threadIdx.x & 31;
  int m = row0 + (lane & 15);
  int hi = (lane >> 4) * 16;
  v16bf a;
#pragma unroll
  for (int i = 0; i < 16; ++i) {
    int k = k0 + hi + i;
    a[i] = s[m * ldk + k];
  }
  return a;
}

// B-matrix 32x16 fragment from row-major (K x N) storage:
// lane L holds col n=L&15; element i -> K index k0 + (L>>4)*16 + i
__device__ __forceinline__ v16bf load_b_frag(const bf16* s, int k0, int col0, int ldn) {
  int lane = threadIdx.x & 31;
  int n = col0 + (lane & 15);
  int hi = (lane >> 4) * 16;
  v16bf b;
#pragma unroll
  for (int i = 0; i < 16; ++i) {
    int k = k0 + hi + i;
    b[i] = s[k * ldn + n];
  }
  return b;
}

// B fragment where storage is (N x K) row-major (i.e. we need B = storage^T)
__device__ __forceinline__ v16bf load_bT_frag(const bf16* s, int k0, int col0, int ld) {
  int lane = threadIdx.x & 31;
  int n = col0 + (lane & 15);
  int hi = (lane >> 4) * 16;
  v16bf b;
#pragma unroll
  for (int i = 0; i < 16; ++i) {
    int k = k0 + hi + i;
    b[i] = s[n * ld + k];
  }
  return b;
}

// ---------------------------------------------------------------------------
// CDNA5 async global->LDS copy (ASYNCcnt-tracked, bypasses VGPRs).
// GVS addressing: SADDR = uniform 64-bit base, VADDR = per-lane 32-bit offset,
// VDST = per-lane LDS byte address. 16 bytes per lane per instruction.
// ---------------------------------------------------------------------------
__device__ __forceinline__ void async_ldg_b128(unsigned lds_addr, const void* base,
                                               int byte_off) {
  asm volatile("global_load_async_to_lds_b128 %0, %1, %2"
               :: "v"(lds_addr), "v"(byte_off), "s"(base)
               : "memory");
}

// Copy `chunks` x 16 bytes from gbase to LDS buffer, spread over 128 threads.
__device__ __forceinline__ void async_copy_tile(const void* gbase, void* lbuf,
                                                int chunks, int tid) {
  unsigned lbase = (unsigned)(unsigned long long)lbuf;
  for (int i = tid; i < chunks; i += 128) {
    async_ldg_b128(lbase + (unsigned)i * 16u, gbase, i * 16);
  }
}

// ---------------------------------------------------------------------------
// Generic tiled WMMA GEMM: C[M x N] = A[M x K] * B[K x N]
//   TRA: A stored transposed (A[m][k] = Aptr[k*lda + m])
//   TRB: B stored transposed (B[k][n] = Bptr[n*ldb + k])
// Block = 128 threads (4 waves). Tile 64x64, k-step 32.
// ---------------------------------------------------------------------------
template <bool TRA, bool TRB, typename TA, typename TB, typename TC>
__global__ __launch_bounds__(128)
void gemm_wmma(const TA* __restrict__ A, const TB* __restrict__ B, TC* __restrict__ C,
               int M, int N, int K, long lda, long ldb, long ldc) {
  __shared__ bf16 As[64 * 32];
  __shared__ bf16 Bs[32 * 64];
  const int wave = threadIdx.x >> 5;
  const int m0 = blockIdx.y * 64;
  const int n0 = blockIdx.x * 64;

  v8f acc[4] = {};

  for (int k0 = 0; k0 < K; k0 += 32) {
    for (int idx = threadIdx.x; idx < 64 * 32; idx += 128) {
      int r = idx >> 5, c = idx & 31;
      float v = TRA ? (float)A[(long)(k0 + c) * lda + (m0 + r)]
                    : (float)A[(long)(m0 + r) * lda + (k0 + c)];
      As[idx] = (bf16)v;
    }
    for (int idx = threadIdx.x; idx < 32 * 64; idx += 128) {
      int r = idx >> 6, c = idx & 63;
      float v = TRB ? (float)B[(long)(n0 + c) * ldb + (k0 + r)]
                    : (float)B[(long)(k0 + r) * ldb + (n0 + c)];
      Bs[idx] = (bf16)v;
    }
    __syncthreads();
    v16bf a = load_a_frag(As, wave * 16, 0, 32);
#pragma unroll
    for (int t = 0; t < 4; ++t) {
      v16bf b = load_b_frag(Bs, 0, t * 16, 64);
      acc[t] = wmma_bf16(a, b, acc[t]);
    }
    __syncthreads();
  }

  const int lane = threadIdx.x & 31;
  const int cn = n0 + (lane & 15);
#pragma unroll
  for (int t = 0; t < 4; ++t) {
#pragma unroll
    for (int r = 0; r < 8; ++r) {
      int rm = m0 + wave * 16 + (lane >> 4) * 8 + r;
      C[(long)rm * ldc + (cn + t * 16)] = (TC)acc[t][r];
    }
  }
}

// ---------------------------------------------------------------------------
// RoPE kernels: fill rotary dims [512..576) of the concatenated Q / K buffers
// ---------------------------------------------------------------------------
__global__ void rope_q_kernel(const float* __restrict__ c_qr,   // (B,T,NH,64)
                              const float* __restrict__ fcos,   // (T,32)
                              const float* __restrict__ fsin,   // (T,32)
                              bf16* __restrict__ Qcat) {        // (B,NH,T,576)
  int idx = blockIdx.x * blockDim.x + threadIdx.x;   // B*T*NH*32 = 2^21
  if (idx >= 4 * 1024 * 16 * 32) return;
  int j = idx & 31;
  int h = (idx >> 5) & 15;
  int t = (idx >> 9) & 1023;
  int b = idx >> 19;
  long src = (((long)(b * 1024 + t) * 16 + h) * 64) + 2 * j;
  float re = c_qr[src], im = c_qr[src + 1];
  float c = fcos[t * 32 + j], s = fsin[t * 32 + j];
  long dst = (((long)(b * 16 + h) * 1024 + t) * 576) + 512 + 2 * j;
  Qcat[dst]     = (bf16)(re * c - im * s);
  Qcat[dst + 1] = (bf16)(re * s + im * c);
}

__global__ void rope_k_kernel(const float* __restrict__ c_kr,   // (B,T,64)
                              const float* __restrict__ fcos,
                              const float* __restrict__ fsin,
                              bf16* __restrict__ Kcat) {        // (B,T,576)
  int idx = blockIdx.x * blockDim.x + threadIdx.x;   // B*T*32 = 131072
  if (idx >= 4 * 1024 * 32) return;
  int j = idx & 31;
  int t = (idx >> 5) & 1023;
  int b = idx >> 15;
  long src = ((long)(b * 1024 + t) * 64) + 2 * j;
  float re = c_kr[src], im = c_kr[src + 1];
  float c = fcos[t * 32 + j], s = fsin[t * 32 + j];
  long dst = ((long)(b * 1024 + t) * 576) + 512 + 2 * j;
  Kcat[dst]     = (bf16)(re * c - im * s);
  Kcat[dst + 1] = (bf16)(re * s + im * c);
}

// ---------------------------------------------------------------------------
// Flash attention core, fused with the output projection.
// Grid: (T/16, NH, B). Block: 128 threads = 4 wave32.
// Double-buffered async K streaming: while WMMAs process key block kb, the
// async unit fills the other LDS buffer with block kb+1 (ASYNCcnt-tracked).
// O accumulator: 16 x 512 f32, wave w owns cols [128w, 128w+128) (8 C-tiles).
// Epilogue: Y(16x128) = O(16x512) @ v_eff[h](512x128), written f32 to d_out.
//
// LDS layout (dynamic, 172736 B total):
//   [0,       18432)  Qs (16x576 bf16)   -- reused as Ob (16x512 bf16) after loop
//   [18432,   92160)  Ks0 (64x576 bf16)
//   [92160,  165888)  Ks1 (64x576 bf16)
//   [165888, 169984)  Ssc (16x64 f32)
//   [169984, 172032)  Ps  (16x64 bf16)
//   [172032, 172544)  red (16x8 f32)
//   [172544, 172736)  m_run/l_run/alpha (16 f32 each)
// ---------------------------------------------------------------------------
__global__ __launch_bounds__(128)
void mla_attention_kernel(const bf16* __restrict__ Qcat,   // (B,NH,T,576)
                          const bf16* __restrict__ Kcat,   // (B,T,576)
                          const bf16* __restrict__ veff,   // (512, 2048) = (NLKV, C)
                          float* __restrict__ out) {       // (B,T,C)
  const float scale = 0.0721687836487032f;  // 1/sqrt(128+64)
  const int t0 = blockIdx.x * 16;
  const int h  = blockIdx.y;
  const int b  = blockIdx.z;
  const int tid  = threadIdx.x;
  const int wave = tid >> 5;
  const int lane = tid & 31;

  extern __shared__ char smem[];
  bf16*  Qs      = (bf16*)smem;
  bf16*  Ob      = (bf16*)smem;
  bf16*  Ks0     = (bf16*)(smem + 18432);
  bf16*  Ks1     = (bf16*)(smem + 92160);
  float* Ssc     = (float*)(smem + 165888);
  bf16*  Ps      = (bf16*)(smem + 169984);
  float* red     = (float*)(smem + 172032);
  float* m_run   = (float*)(smem + 172544);
  float* l_run   = m_run + 16;
  float* alpha_s = l_run + 16;

  const bf16* Qbase = Qcat + ((long)(b * 16 + h) * 1024 + t0) * 576;
  const bf16* Kbat  = Kcat + (long)b * 1024 * 576;

  // Prologue: async-copy Q tile (1152 x 16B) and key block 0 (4608 x 16B).
  async_copy_tile(Qbase, Qs, 1152, tid);
  async_copy_tile(Kbat, Ks0, 4608, tid);

  if (tid < 16) { m_run[tid] = -INFINITY; l_run[tid] = 0.0f; }
  v8f acc[8] = {};

  const int nkb = t0 / 64 + 1;  // causal: key blocks s0=0..t0
  for (int kb = 0; kb < nkb; ++kb) {
    const int s0 = kb * 64;
    bf16* Kb = (kb & 1) ? Ks1 : Ks0;

    if (kb + 1 < nkb) {
      // Issue next key block into the other buffer, then wait until only the
      // 36 per-wave transfers just issued remain outstanding (in-order
      // completion => everything issued earlier, incl. current block, landed).
      bf16* Kn = (kb & 1) ? Ks0 : Ks1;
      async_copy_tile(Kbat + (long)(kb + 1) * 64 * 576, Kn, 4608, tid);
      asm volatile("s_wait_asynccnt 0x24" ::: "memory");
    } else {
      asm volatile("s_wait_asynccnt 0x0" ::: "memory");
    }
    __syncthreads();

    // S tile: wave w computes key cols [16w, 16w+16); 576 = 18 k-steps of 32
    v8f s = {};
    for (int ks = 0; ks < 18; ++ks) {
      v16bf a  = load_a_frag(Qs, 0, ks * 32, 576);
      v16bf bt = load_bT_frag(Kb, ks * 32, wave * 16, 576);
      s = wmma_bf16(a, bt, s);
    }
    {
      int n = lane & 15;
      int jcol = wave * 16 + n;
      int sabs = s0 + jcol;
#pragma unroll
      for (int r = 0; r < 8; ++r) {
        int row = (lane >> 4) * 8 + r;
        float v = s[r] * scale;
        if (sabs > t0 + row) v = -INFINITY;   // causal mask
        Ssc[row * 64 + jcol] = v;
      }
    }
    __syncthreads();

    // Online softmax, parallelized: 8 threads per query row, 8 cols each.
    {
      int r = tid >> 3, g = tid & 7;
      float mx = -INFINITY;
#pragma unroll
      for (int j = 0; j < 8; ++j) mx = fmaxf(mx, Ssc[r * 64 + g * 8 + j]);
      red[r * 8 + g] = mx;
    }
    __syncthreads();
    if (tid < 16) {
      float m_old = m_run[tid];
      float mx = m_old;
#pragma unroll
      for (int g = 0; g < 8; ++g) mx = fmaxf(mx, red[tid * 8 + g]);
      alpha_s[tid] = (m_old == -INFINITY) ? 0.0f : __expf(m_old - mx);
      m_run[tid] = mx;
    }
    __syncthreads();
    {
      int r = tid >> 3, g = tid & 7;
      float mx = m_run[r];
      float rs = 0.0f;
#pragma unroll
      for (int j = 0; j < 8; ++j) {
        float p = __expf(Ssc[r * 64 + g * 8 + j] - mx);
        rs += p;
        Ps[r * 64 + g * 8 + j] = (bf16)p;
      }
      red[r * 8 + g] = rs;
    }
    __syncthreads();
    if (tid < 16) {
      float rs = 0.0f;
#pragma unroll
      for (int g = 0; g < 8; ++g) rs += red[tid * 8 + g];
      l_run[tid] = l_run[tid] * alpha_s[tid] + rs;
    }

    // Rescale accumulators, then O += P(16x64) @ V(64x512); V = Kb[:, :512]
#pragma unroll
    for (int t = 0; t < 8; ++t) {
#pragma unroll
      for (int r = 0; r < 8; ++r) {
        int row = (lane >> 4) * 8 + r;
        acc[t][r] *= alpha_s[row];
      }
    }
    for (int ks = 0; ks < 2; ++ks) {
      v16bf a = load_a_frag(Ps, 0, ks * 32, 64);
#pragma unroll
      for (int t = 0; t < 8; ++t) {
        int c0 = wave * 128 + t * 16;
        v16bf bv = load_b_frag(Kb, ks * 32, c0, 576);
        acc[t] = wmma_bf16(a, bv, acc[t]);
      }
    }
    __syncthreads();  // all compute on Kb done before next issue overwrites
  }

  // Normalize and stash O as bf16 (reuse Qs storage)
#pragma unroll
  for (int t = 0; t < 8; ++t) {
    int c0 = wave * 128 + t * 16 + (lane & 15);
#pragma unroll
    for (int r = 0; r < 8; ++r) {
      int row = (lane >> 4) * 8 + r;
      Ob[row * 512 + c0] = (bf16)(acc[t][r] / l_run[row]);
    }
  }
  __syncthreads();

  // Epilogue: Y(16x128) = Ob(16x512) @ veff[:, h*128 : h*128+128]
  v8f y[2] = {};
  const bf16* Vh = veff + (long)h * 128;  // row stride 2048
  for (int ks = 0; ks < 16; ++ks) {
    v16bf a = load_a_frag(Ob, 0, ks * 32, 512);
#pragma unroll
    for (int t = 0; t < 2; ++t) {
      v16bf bv = load_b_frag(Vh, ks * 32, wave * 32 + t * 16, 2048);
      y[t] = wmma_bf16(a, bv, y[t]);
    }
  }
  {
    int n = lane & 15;
#pragma unroll
    for (int t = 0; t < 2; ++t) {
      int col = wave * 32 + t * 16 + n;
#pragma unroll
      for (int r = 0; r < 8; ++r) {
        int row = (lane >> 4) * 8 + r;
        out[(long)(b * 1024 + t0 + row) * 2048 + h * 128 + col] = y[t][r];
      }
    }
  }
}

// ---------------------------------------------------------------------------
// Host-side orchestration
// ---------------------------------------------------------------------------
extern "C" void kernel_launch(void* const* d_in, const int* in_sizes, int n_in,
                              void* d_out, int out_size, void* d_ws, size_t ws_size,
                              hipStream_t stream) {
  (void)in_sizes; (void)n_in; (void)out_size; (void)ws_size;
  const float* x     = (const float*)d_in[0];   // (4,1024,2048)
  const float* W_dq  = (const float*)d_in[1];   // (512,2048)
  const float* W_uq  = (const float*)d_in[2];   // (2048,512) ~ flat (512,16,128)
  const float* W_dkv = (const float*)d_in[3];   // (512,2048)
  const float* W_uk  = (const float*)d_in[4];   // (2048,512) ~ flat (16,128,512)
  const float* W_uv  = (const float*)d_in[5];   // (2048,512)
  const float* W_o   = (const float*)d_in[6];   // (2048,2048)
  const float* W_qr  = (const float*)d_in[7];   // (1024,512)
  const float* W_kr  = (const float*)d_in[8];   // (64,2048)
  const float* fcos  = (const float*)d_in[9];   // (1024,32)
  const float* fsin  = (const float*)d_in[10];  // (1024,32)
  float* out = (float*)d_out;

  char* ws = (char*)d_ws;
  size_t off = 0;
  auto take = [&](size_t bytes) -> char* {
    char* p = ws + off;
    off += (bytes + 255) & ~(size_t)255;
    return p;
  };
  float* c_q  = (float*)take((size_t)4096 * 512 * 4);          // (B*T, 512)
  float* c_qr = (float*)take((size_t)4096 * 1024 * 4);         // (B*T, NH*64)
  float* c_kr = (float*)take((size_t)4096 * 64 * 4);           // (B*T, 64)
  bf16*  keff = (bf16*)take((size_t)16 * 512 * 512 * 2);       // (NH, 512, 512)
  bf16*  veff = (bf16*)take((size_t)512 * 2048 * 2);           // (512, 2048)
  bf16*  Qcat = (bf16*)take((size_t)64 * 1024 * 576 * 2);      // (B,NH,T,576)
  bf16*  Kcat = (bf16*)take((size_t)4 * 1024 * 576 * 2);       // (B,T,576)

  const dim3 blk(128);

  // c_q = x @ W_dq^T            (4096 x 512, K=2048)
  gemm_wmma<false, true, float, float, float>
      <<<dim3(8, 64), blk, 0, stream>>>(x, W_dq, c_q, 4096, 512, 2048, 2048, 2048, 512);

  // Kcat[:, :512] = x @ W_dkv^T (bf16)
  gemm_wmma<false, true, float, float, bf16>
      <<<dim3(8, 64), blk, 0, stream>>>(x, W_dkv, Kcat, 4096, 512, 2048, 2048, 2048, 576);

  // c_kr = x @ W_kr^T           (4096 x 64)
  gemm_wmma<false, true, float, float, float>
      <<<dim3(1, 64), blk, 0, stream>>>(x, W_kr, c_kr, 4096, 64, 2048, 2048, 2048, 64);

  // k_eff[h] = W_uq[:,h,:] (512x128) @ W_uk[h] (128x512), bf16
  for (int h = 0; h < 16; ++h) {
    gemm_wmma<false, false, float, float, bf16>
        <<<dim3(8, 8), blk, 0, stream>>>(W_uq + (long)h * 128, W_uk + (long)h * 65536,
                                         keff + (long)h * 262144,
                                         512, 512, 128, 2048, 512, 512);
  }

  // v_eff = W_uv^T @ W_o^T      (512 x 2048, K=2048), bf16
  gemm_wmma<true, true, float, float, bf16>
      <<<dim3(32, 8), blk, 0, stream>>>(W_uv, W_o, veff, 512, 2048, 2048, 512, 2048, 2048);

  // c_qr = c_q @ W_qr^T         (4096 x 1024, K=512)
  gemm_wmma<false, true, float, float, float>
      <<<dim3(16, 64), blk, 0, stream>>>(c_q, W_qr, c_qr, 4096, 1024, 512, 512, 512, 1024);

  // Qcat[b,h,:, :512] = c_q[b] @ k_eff[h]  (1024 x 512, K=512), bf16
  for (int b = 0; b < 4; ++b) {
    for (int h = 0; h < 16; ++h) {
      gemm_wmma<false, false, float, bf16, bf16>
          <<<dim3(8, 16), blk, 0, stream>>>(c_q + (long)b * 1024 * 512,
                                            keff + (long)h * 262144,
                                            Qcat + ((long)(b * 16 + h) * 1024) * 576,
                                            1024, 512, 512, 512, 512, 576);
    }
  }

  // RoPE fills cols [512..576)
  rope_q_kernel<<<dim3((4 * 1024 * 16 * 32) / 256), dim3(256), 0, stream>>>(c_qr, fcos, fsin, Qcat);
  rope_k_kernel<<<dim3((4 * 1024 * 32) / 256), dim3(256), 0, stream>>>(c_kr, fcos, fsin, Kcat);

  // Flash attention + fused output projection (double-buffered async K stream)
  const size_t smem_bytes = 172736;
  mla_attention_kernel<<<dim3(64, 16, 4), blk, smem_bytes, stream>>>(Qcat, Kcat, veff, out);
}
